// LadderEncoder_67697274519897
// MI455X (gfx1250) — matrix operand order repaired and checked by reading, not compiled, not run on hardware
//
#include <hip/hip_runtime.h>
#include <math.h>

// ---------------------------------------------------------------------------
// Types for CDNA5 WMMA (wave32): v_wmma_f32_16x16x32_bf16
// ---------------------------------------------------------------------------
typedef __bf16 v16bf __attribute__((ext_vector_type(16)));
typedef float  v8f   __attribute__((ext_vector_type(8)));
typedef unsigned int v4u __attribute__((ext_vector_type(4)));

union AFrag { v16bf v; v4u u[2]; };
union CFrag { v8f   v; float f[8]; };

// bf16 (round-to-nearest-even) <-> f32 via integer ops (no __bf16 arithmetic).
static __device__ __forceinline__ unsigned short f2bf(float f) {
    unsigned int x = __float_as_uint(f);
    x += 0x7FFFu + ((x >> 16) & 1u);
    return (unsigned short)(x >> 16);
}
static __device__ __forceinline__ float bf2f(unsigned short h) {
    return __uint_as_float(((unsigned int)h) << 16);
}

// Problem constants
#define NE      4032        // edges = N*(N-1)
#define ME      258048L     // (B*S) * NE
#define MN      4096L       // (B*S) * N
#define OUT_MZ  0L
#define OUT_VZ  258048L
#define OUT_MB  516096L
#define OUT_VB  532224L

// ---------------------------------------------------------------------------
// Weight packing: W [K x 256] f32 -> fragment-native bf16 blobs.
// Packed index: (((kb*16 + nb)*32 + lane)*16 + e)
// lane: col = nb*16 + (lane&15);  K = kb*32 + (lane>>4)*8 + (e&7) + (e>>3)*16
// -> each lane's 16 B-values are 32 contiguous bytes (two b128 loads).
// ---------------------------------------------------------------------------
__global__ void pack_w_kernel(const float* __restrict__ w,
                              unsigned short* __restrict__ p,
                              int kreal, int total) {
    int idx = blockIdx.x * 256 + threadIdx.x;
    if (idx >= total) return;
    int e  = idx & 15;
    int L  = (idx >> 4) & 31;
    int nb = (idx >> 9) & 15;
    int kb = idx >> 13;
    int k  = kb * 32 + (L >> 4) * 8 + (e & 7) + (e >> 3) * 16;
    int n  = nb * 16 + (L & 15);
    float v = (k < kreal) ? w[(long)k * 256 + n] : 0.0f;
    p[idx] = f2bf(v);
}

// ---------------------------------------------------------------------------
// data [B,S,T,N,F] -> x_in bf16 [4096, 96]  (K = T*F = 80, zero-padded to 96)
// ---------------------------------------------------------------------------
__global__ void build_xin_kernel(const float* __restrict__ data,
                                 unsigned short* __restrict__ xin) {
    long idx = (long)blockIdx.x * 256 + threadIdx.x;
    if (idx >= MN * 96) return;
    int  k    = (int)(idx % 96);
    long r    = idx / 96;
    int  node = (int)(r & 63);
    long bsi  = r >> 6;
    float v = 0.0f;
    if (k < 80) {
        int t = k >> 2, f = k & 3;
        v = data[((bsi * 20 + t) * 64 + node) * 4 + f];
    }
    xin[idx] = f2bf(v);
}

// ---------------------------------------------------------------------------
// edge2node: node r sums its 63 contiguous incoming edges, / 64.
// ---------------------------------------------------------------------------
__global__ void edge2node_kernel(const unsigned short* __restrict__ xskip,
                                 unsigned short* __restrict__ nodes_in) {
    int col  = threadIdx.x;      // 0..255
    int rowz = blockIdx.x;       // 0..4095
    int  r  = rowz & 63;
    long bs = rowz >> 6;
    const unsigned short* base = xskip + (bs * NE + (long)r * 63) * 256 + col;
    float s = 0.0f;
    for (int c = 0; c < 63; ++c) s += bf2f(base[(long)c * 256]);
    nodes_in[(long)rowz * 256 + col] = f2bf(s * (1.0f / 64.0f));
}

// ---------------------------------------------------------------------------
// Fused 2-layer MLP (GEMM + bias + ELU + GEMM + bias) via WMMA bf16.
// Tile: 32 rows x 256 cols, 256 threads = 8 waves.
//   wave w owns N-blocks {2w, 2w+1}; both 16-row M-blocks share its B frags,
//   so each k-step is 4 WMMAs per (2 A-frag + 2 B-frag) b128 load pairs.
// KB1 is a template constant -> layer-1 loop fully unrolled, chunk selection
// static (no indexed pointer array, no movrels, global_load addressing).
// MODE 0: A = plain bf16 matrix [M, lda]
// MODE 1: A row m = concat(x1[recv(m)], x1[send(m)])              (K=512)
// MODE 2: A row m = concat(nodes[recv], nodes[send], x_skip[m])   (K=768)
// HEAD  : compute mu / softplus-var heads into d_out instead of storing h.
// ---------------------------------------------------------------------------
template <int MODE, bool HEAD, int KB1>
__global__ void fused_mlp_kernel(const unsigned short* __restrict__ A,
                                 const unsigned short* __restrict__ A2,
                                 int lda,
                                 const unsigned short* __restrict__ W1p,
                                 const float* __restrict__ b1,
                                 const unsigned short* __restrict__ W2p,
                                 const float* __restrict__ b2,
                                 unsigned short* __restrict__ Out,
                                 const float* __restrict__ wm,
                                 const float* __restrict__ bm,
                                 const float* __restrict__ wv,
                                 const float* __restrict__ bv,
                                 float* __restrict__ mean_z,
                                 float* __restrict__ var_z) {
    const int tid  = threadIdx.x;
    const int wave = tid >> 5;
    const int lane = tid & 31;
    const int hi   = lane >> 4;
    const int lo   = lane & 15;
    const long m0  = (long)blockIdx.x * 32;

    // Per-lane source chunk pointers for the two 16-row M-blocks.
    // (Statically selected below since KB1 is a compile-time constant.)
    const unsigned short* cp00; const unsigned short* cp01; const unsigned short* cp02;
    const unsigned short* cp10; const unsigned short* cp11; const unsigned short* cp12;
    {
        const long mA = m0 + lo;
        const long mB = m0 + 16 + lo;
        if constexpr (MODE == 0) {
            cp00 = A + mA * (long)lda; cp01 = cp00; cp02 = cp00;
            cp10 = A + mB * (long)lda; cp11 = cp10; cp12 = cp10;
        } else {
            {
                long bs = mA / NE; long ei = mA - bs * NE;
                int i = (int)(ei / 63); int jj = (int)(ei - (long)i * 63);
                int j = jj + (jj >= i ? 1 : 0);
                cp00 = A + ((bs << 6) + i) * 256L;
                cp01 = A + ((bs << 6) + j) * 256L;
                cp02 = (MODE == 2) ? (A2 + mA * 256L) : cp00;
            }
            {
                long bs = mB / NE; long ei = mB - bs * NE;
                int i = (int)(ei / 63); int jj = (int)(ei - (long)i * 63);
                int j = jj + (jj >= i ? 1 : 0);
                cp10 = A + ((bs << 6) + i) * 256L;
                cp11 = A + ((bs << 6) + j) * 256L;
                cp12 = (MODE == 2) ? (A2 + mB * 256L) : cp10;
            }
        }
    }

    __shared__ __align__(16) unsigned short s_hid[32 * 256];

    const int nb0 = wave * 2;
    const int nb1 = nb0 + 1;

    CFrag acc[2][2];   // [m-block][n-block]
#pragma unroll
    for (int mb = 0; mb < 2; ++mb)
#pragma unroll
        for (int v = 0; v < 8; ++v) { acc[mb][0].f[v] = 0.0f; acc[mb][1].f[v] = 0.0f; }

    // ---------------- Layer 1: hidden = A @ W1 ----------------
#pragma unroll
    for (int kb = 0; kb < KB1; ++kb) {
        const int k0 = kb * 32;
        const int ch = k0 >> 8;          // compile-time constant per iteration
        const int kl = k0 & 255;
        const unsigned short* p0 = (ch == 0) ? cp00 : (ch == 1) ? cp01 : cp02;
        const unsigned short* p1 = (ch == 0) ? cp10 : (ch == 1) ? cp11 : cp12;
        AFrag a0, a1;
        a0.u[0] = *(const v4u*)(p0 + kl + hi * 8);
        a0.u[1] = *(const v4u*)(p0 + kl + hi * 8 + 16);
        a1.u[0] = *(const v4u*)(p1 + kl + hi * 8);
        a1.u[1] = *(const v4u*)(p1 + kl + hi * 8 + 16);
        const unsigned short* bp0 = W1p + (((long)kb * 16 + nb0) * 32 + lane) * 16;
        const unsigned short* bp1 = W1p + (((long)kb * 16 + nb1) * 32 + lane) * 16;
        AFrag bA, bB;
        bA.u[0] = *(const v4u*)(bp0);  bA.u[1] = *(const v4u*)(bp0 + 8);
        bB.u[0] = *(const v4u*)(bp1);  bB.u[1] = *(const v4u*)(bp1 + 8);
        acc[0][0].v = __builtin_amdgcn_wmma_f32_16x16x32_bf16(false, a0.v, false, bA.v,
                                                              (short)0, acc[0][0].v, false, false);
        acc[0][1].v = __builtin_amdgcn_wmma_f32_16x16x32_bf16(false, a0.v, false, bB.v,
                                                              (short)0, acc[0][1].v, false, false);
        acc[1][0].v = __builtin_amdgcn_wmma_f32_16x16x32_bf16(false, a1.v, false, bA.v,
                                                              (short)0, acc[1][0].v, false, false);
        acc[1][1].v = __builtin_amdgcn_wmma_f32_16x16x32_bf16(false, a1.v, false, bB.v,
                                                              (short)0, acc[1][1].v, false, false);
    }

    // bias + ELU -> bf16 hidden tile in LDS
    {
        const float bb0 = b1[nb0 * 16 + lo];
        const float bb1 = b1[nb1 * 16 + lo];
#pragma unroll
        for (int mb = 0; mb < 2; ++mb) {
#pragma unroll
            for (int v = 0; v < 8; ++v) {
                const int r = mb * 16 + v + 8 * hi;
                float h0 = acc[mb][0].f[v] + bb0; h0 = (h0 > 0.0f) ? h0 : expm1f(h0);
                float h1 = acc[mb][1].f[v] + bb1; h1 = (h1 > 0.0f) ? h1 : expm1f(h1);
                s_hid[r * 256 + nb0 * 16 + lo] = f2bf(h0);
                s_hid[r * 256 + nb1 * 16 + lo] = f2bf(h1);
            }
        }
    }
    __syncthreads();

    // ---------------- Layer 2: out = hidden @ W2 ----------------
#pragma unroll
    for (int mb = 0; mb < 2; ++mb)
#pragma unroll
        for (int v = 0; v < 8; ++v) { acc[mb][0].f[v] = 0.0f; acc[mb][1].f[v] = 0.0f; }
#pragma unroll
    for (int kb = 0; kb < 8; ++kb) {
        const int k0 = kb * 32;
        AFrag a0, a1;
        a0.u[0] = *(const v4u*)(&s_hid[(lo)      * 256 + k0 + hi * 8]);
        a0.u[1] = *(const v4u*)(&s_hid[(lo)      * 256 + k0 + hi * 8 + 16]);
        a1.u[0] = *(const v4u*)(&s_hid[(16 + lo) * 256 + k0 + hi * 8]);
        a1.u[1] = *(const v4u*)(&s_hid[(16 + lo) * 256 + k0 + hi * 8 + 16]);
        const unsigned short* bp0 = W2p + (((long)kb * 16 + nb0) * 32 + lane) * 16;
        const unsigned short* bp1 = W2p + (((long)kb * 16 + nb1) * 32 + lane) * 16;
        AFrag bA, bB;
        bA.u[0] = *(const v4u*)(bp0);  bA.u[1] = *(const v4u*)(bp0 + 8);
        bB.u[0] = *(const v4u*)(bp1);  bB.u[1] = *(const v4u*)(bp1 + 8);
        acc[0][0].v = __builtin_amdgcn_wmma_f32_16x16x32_bf16(false, a0.v, false, bA.v,
                                                              (short)0, acc[0][0].v, false, false);
        acc[0][1].v = __builtin_amdgcn_wmma_f32_16x16x32_bf16(false, a0.v, false, bB.v,
                                                              (short)0, acc[0][1].v, false, false);
        acc[1][0].v = __builtin_amdgcn_wmma_f32_16x16x32_bf16(false, a1.v, false, bA.v,
                                                              (short)0, acc[1][0].v, false, false);
        acc[1][1].v = __builtin_amdgcn_wmma_f32_16x16x32_bf16(false, a1.v, false, bB.v,
                                                              (short)0, acc[1][1].v, false, false);
    }

    const float bb0 = b2[nb0 * 16 + lo];
    const float bb1 = b2[nb1 * 16 + lo];

    if constexpr (!HEAD) {
        // Store bf16 activations for the next stage.
#pragma unroll
        for (int mb = 0; mb < 2; ++mb) {
#pragma unroll
            for (int v = 0; v < 8; ++v) {
                const long row = m0 + mb * 16 + v + 8 * hi;
                Out[row * 256 + nb0 * 16 + lo] = f2bf(acc[mb][0].f[v] + bb0);
                Out[row * 256 + nb1 * 16 + lo] = f2bf(acc[mb][1].f[v] + bb1);
            }
        }
    } else {
        // Fused mu / var heads: h tile (f32) -> LDS -> per-row dot products.
        __shared__ float s_hf[32 * 256];
#pragma unroll
        for (int mb = 0; mb < 2; ++mb) {
#pragma unroll
            for (int v = 0; v < 8; ++v) {
                const int r = mb * 16 + v + 8 * hi;
                s_hf[r * 256 + nb0 * 16 + lo] = acc[mb][0].f[v] + bb0;
                s_hf[r * 256 + nb1 * 16 + lo] = acc[mb][1].f[v] + bb1;
            }
        }
        __syncthreads();
        if (tid < 64) {
            const int r   = tid & 31;
            const int sel = tid >> 5;          // 0 = mu, 1 = var
            const long row = m0 + r;
            const float* w = sel ? wv : wm;
            float accv = sel ? bv[0] : bm[0];
            for (int k = 0; k < 256; ++k) accv += s_hf[r * 256 + k] * w[k];
            if (sel == 0) {
                mean_z[row] = accv;
            } else {
                float sp = (accv > 20.0f) ? accv : log1pf(expf(accv));
                sp = fminf(fmaxf(sp, 1e-8f), 100.0f);
                var_z[row] = sp;
            }
        }
    }
}

// ---------------------------------------------------------------------------
// Entity2CommonEncoder: per (b, e) reduce over S=16 subjects.
//   mu_bar = S*mean/(S+var),  var_bar = clip(var/(S+var))
// ---------------------------------------------------------------------------
__global__ void entity2common_kernel(const float* __restrict__ mean_z,
                                     float* __restrict__ mu_bar,
                                     float* __restrict__ var_bar) {
    int t = blockIdx.x * 256 + threadIdx.x;
    if (t >= 4 * NE) return;
    int b = t / NE;
    int e = t - b * NE;
    float xs[16];
    float mean = 0.0f;
#pragma unroll
    for (int s = 0; s < 16; ++s) {
        xs[s] = mean_z[((long)(b * 16 + s)) * NE + e];
        mean += xs[s];
    }
    mean *= (1.0f / 16.0f);
    float var = 0.0f;
#pragma unroll
    for (int s = 0; s < 16; ++s) { float d = xs[s] - mean; var += d * d; }
    var *= (1.0f / 16.0f);
    mu_bar[t] = 16.0f * mean / (16.0f + var);
    float vb = var / (16.0f + var);
    var_bar[t] = fminf(fmaxf(vb, 1e-8f), 100.0f);
}

// ---------------------------------------------------------------------------
// Host launch
// ---------------------------------------------------------------------------
extern "C" void kernel_launch(void* const* d_in, const int* in_sizes, int n_in,
                              void* d_out, int out_size, void* d_ws, size_t ws_size,
                              hipStream_t stream) {
    const float* data = (const float*)d_in[0];
    const float* w1a = (const float*)d_in[1];  const float* b1a = (const float*)d_in[2];
    const float* w1b = (const float*)d_in[3];  const float* b1b = (const float*)d_in[4];
    const float* w2a = (const float*)d_in[5];  const float* b2a = (const float*)d_in[6];
    const float* w2b = (const float*)d_in[7];  const float* b2b = (const float*)d_in[8];
    const float* w3a = (const float*)d_in[9];  const float* b3a = (const float*)d_in[10];
    const float* w3b = (const float*)d_in[11]; const float* b3b = (const float*)d_in[12];
    const float* wha = (const float*)d_in[13]; const float* bha = (const float*)d_in[14];
    const float* whb = (const float*)d_in[15]; const float* bhb = (const float*)d_in[16];
    const float* wm  = (const float*)d_in[17]; const float* bm  = (const float*)d_in[18];
    const float* wv  = (const float*)d_in[19]; const float* bv  = (const float*)d_in[20];
    (void)in_sizes; (void)n_in; (void)out_size; (void)ws_size;

    float* out = (float*)d_out;
    unsigned short* ws = (unsigned short*)d_ws;
    size_t off = 0;
    auto take = [&](size_t n) { unsigned short* p = ws + off; off += n; return p; };

    // Packed weights (bf16 fragment layout): KB*16*512 elements each.
    unsigned short* w1a_p = take(3  * 8192);
    unsigned short* w1b_p = take(8  * 8192);
    unsigned short* w2a_p = take(16 * 8192);
    unsigned short* w2b_p = take(8  * 8192);
    unsigned short* w3a_p = take(8  * 8192);
    unsigned short* w3b_p = take(8  * 8192);
    unsigned short* wha_p = take(24 * 8192);
    unsigned short* whb_p = take(8  * 8192);
    // Activations (bf16).
    unsigned short* x_in     = take(MN * 96);    // [4096, 96]
    unsigned short* x1       = take(MN * 256);   // node embeddings after MLP1
    unsigned short* nodes_in = take(MN * 256);   // aggregated edges / N
    unsigned short* nodes    = take(MN * 256);   // after MLP3
    unsigned short* x_skip   = take(ME * 256);   // edge embeddings after MLP2

    auto pack = [&](const float* w, unsigned short* p, int kreal, int kb) {
        int total = kb * 8192;
        pack_w_kernel<<<(total + 255) / 256, 256, 0, stream>>>(w, p, kreal, total);
    };
    pack(w1a, w1a_p, 80,  3);
    pack(w1b, w1b_p, 256, 8);
    pack(w2a, w2a_p, 512, 16);
    pack(w2b, w2b_p, 256, 8);
    pack(w3a, w3a_p, 256, 8);
    pack(w3b, w3b_p, 256, 8);
    pack(wha, wha_p, 768, 24);
    pack(whb, whb_p, 256, 8);

    build_xin_kernel<<<(int)((MN * 96 + 255) / 256), 256, 0, stream>>>(data, x_in);

    // MLP1: [4096 x 96(pad 80)] -> 256 -> 256
    fused_mlp_kernel<0, false, 3><<<(int)(MN / 32), 256, 0, stream>>>(
        x_in, nullptr, 96, w1a_p, b1a, w1b_p, b1b, x1,
        nullptr, nullptr, nullptr, nullptr, nullptr, nullptr);

    // MLP2 on edges (node2edge gather fused): [258048 x 512] -> 256 -> 256
    fused_mlp_kernel<1, false, 16><<<(int)(ME / 32), 256, 0, stream>>>(
        x1, nullptr, 0, w2a_p, b2a, w2b_p, b2b, x_skip,
        nullptr, nullptr, nullptr, nullptr, nullptr, nullptr);

    // edge2node (contiguous 63-edge sums, / 64)
    edge2node_kernel<<<(int)MN, 256, 0, stream>>>(x_skip, nodes_in);

    // MLP3 on nodes: [4096 x 256] -> 256 -> 256
    fused_mlp_kernel<0, false, 8><<<(int)(MN / 32), 256, 0, stream>>>(
        nodes_in, nullptr, 256, w3a_p, b3a, w3b_p, b3b, nodes,
        nullptr, nullptr, nullptr, nullptr, nullptr, nullptr);

    // MLPh on edges (3-way concat gather fused) + fused mu/var heads.
    fused_mlp_kernel<2, true, 24><<<(int)(ME / 32), 256, 0, stream>>>(
        nodes, x_skip, 0, wha_p, bha, whb_p, bhb, nullptr,
        wm, bm, wv, bv, out + OUT_MZ, out + OUT_VZ);

    // Cross-subject reduction -> mu_bar, var_bar.
    entity2common_kernel<<<(4 * NE + 255) / 256, 256, 0, stream>>>(
        out + OUT_MZ, out + OUT_MB, out + OUT_VB);
}
